// TransformerBlock_28982439313843
// MI455X (gfx1250) — compile-verified, matrix operand order checked by reading, not run on hardware
//
#include <hip/hip_runtime.h>
#include <cstdint>
#include <cstddef>

#define EMBED   1024
#define NHEADS  16
#define HDIM    64
#define HID     4096
#define BATCH   2
#define SEQ     2048
#define MROWS   (BATCH * SEQ)   // 4096

typedef __bf16 bf16;
typedef __attribute__((ext_vector_type(16))) __bf16 bf16x16;
typedef __attribute__((ext_vector_type(8)))  float  f32x8;
typedef __attribute__((ext_vector_type(4)))  int    i32x4;

union FragBF {
    uint4   q[2];
    bf16x16 v;
};

static __device__ inline f32x8 zero8() {
    f32x8 z;
#pragma unroll
    for (int i = 0; i < 8; ++i) z[i] = 0.0f;
    return z;
}

static __device__ inline f32x8 wmma_bf16(const FragBF& a, const FragBF& b, f32x8 c) {
    return __builtin_amdgcn_wmma_f32_16x16x32_bf16(
        /*neg_a=*/false, a.v, /*neg_b=*/false, b.v,
        /*c_mod=*/(short)0, c, /*reuse_a=*/false, /*reuse_b=*/false);
}

// ---------------------------------------------------------------------------
// CDNA5 async global->LDS copy (ASYNCcnt-tracked DMA, bypasses VGPRs).
// Builtin prototype: (v4i32 as1* gaddr, v4i32 as3* lds, imm offset, imm cpol).
// Falls back to a plain b128 round-trip if the builtin isn't exposed.
// ---------------------------------------------------------------------------
#if defined(__AMDGCN__) && __has_builtin(__builtin_amdgcn_global_load_async_to_lds_b128)
#define USE_ASYNC_LDS 1
typedef __attribute__((address_space(1))) i32x4 gas_i32x4;
typedef __attribute__((address_space(3))) i32x4 las_i32x4;
#endif

static __device__ inline void copy16_g2l(const bf16* g, bf16* l) {
#if defined(USE_ASYNC_LDS)
    __builtin_amdgcn_global_load_async_to_lds_b128(
        (gas_i32x4*)(g), (las_i32x4*)(l), 0, 0);
#else
    *reinterpret_cast<uint4*>(l) = *reinterpret_cast<const uint4*>(g);
#endif
}

static __device__ inline void wait_async_lds() {
#if defined(USE_ASYNC_LDS)
#if __has_builtin(__builtin_amdgcn_s_wait_asynccnt)
    __builtin_amdgcn_s_wait_asynccnt(0);
#else
    asm volatile("s_wait_asynccnt 0x0" ::: "memory");
#endif
#endif
}

// ---------------------------------------------------------------------------
// fp32 [K,N] -> bf16 [N,K] transposed conversion (weights, once per launch).
// 32x32 tiles through LDS; block (32,8).
// ---------------------------------------------------------------------------
__global__ void __launch_bounds__(256) cvt_transpose_bf16(
    const float* __restrict__ in, bf16* __restrict__ out, int K, int N)
{
    __shared__ float t[32][33];
    const int nt = blockIdx.x * 32, kt = blockIdx.y * 32;
    const int tx = threadIdx.x, ty = threadIdx.y;
#pragma unroll
    for (int r = 0; r < 32; r += 8)
        t[ty + r][tx] = in[(size_t)(kt + ty + r) * N + nt + tx];
    __syncthreads();
#pragma unroll
    for (int r = 0; r < 32; r += 8)
        out[(size_t)(nt + ty + r) * K + kt + tx] = (bf16)t[tx][ty + r];
}

// ---------------------------------------------------------------------------
// LayerNorm over D=1024, one row per 256-thread block, bf16 output
// ---------------------------------------------------------------------------
__global__ void __launch_bounds__(256) layernorm_bf16(
    const float* __restrict__ x, const float* __restrict__ g,
    const float* __restrict__ b, bf16* __restrict__ out)
{
    __shared__ float red[16];
    const int row = blockIdx.x;
    const int tid = threadIdx.x;
    const float4 v = reinterpret_cast<const float4*>(x + (size_t)row * EMBED)[tid];
    float s  = v.x + v.y + v.z + v.w;
    float s2 = v.x * v.x + v.y * v.y + v.z * v.z + v.w * v.w;
#pragma unroll
    for (int m = 1; m < 32; m <<= 1) {
        s  += __shfl_xor(s,  m, 32);
        s2 += __shfl_xor(s2, m, 32);
    }
    const int w = tid >> 5;
    if ((tid & 31) == 0) { red[w] = s; red[8 + w] = s2; }
    __syncthreads();
    float ts = 0.f, ts2 = 0.f;
#pragma unroll
    for (int i = 0; i < 8; ++i) { ts += red[i]; ts2 += red[8 + i]; }
    const float mu  = ts * (1.0f / EMBED);
    const float var = ts2 * (1.0f / EMBED) - mu * mu;
    const float rs  = rsqrtf(var + 1e-5f);
    const float4 gv = reinterpret_cast<const float4*>(g)[tid];
    const float4 bv = reinterpret_cast<const float4*>(b)[tid];
    bf16* o = out + (size_t)row * EMBED + tid * 4;
    o[0] = (bf16)((v.x - mu) * rs * gv.x + bv.x);
    o[1] = (bf16)((v.y - mu) * rs * gv.y + bv.y);
    o[2] = (bf16)((v.z - mu) * rs * gv.z + bv.z);
    o[3] = (bf16)((v.w - mu) * rs * gv.w + bv.w);
}

// ---------------------------------------------------------------------------
// 128x128-tile bf16 WMMA GEMM, fp32 accumulate, double-buffered LDS with
// async global->LDS tile DMA.  B is pre-transposed: Bt[N,K] row-major, so
// both tiles are straight 128x32 row-major copies.
//   mode 0: outb = (gelu?)(A*Bt^T + bias)     (bf16)
//   mode 1: outf = res + A*Bt^T + bias        (f32 residual add)
//   mode 2: scatter QKV -> Q[B,H,N,Hd], K[B,H,N,Hd], Vt[B,H,Hd,N]  (bf16)
// M,N multiples of 128; K multiple of 32.
// ---------------------------------------------------------------------------
__global__ void __launch_bounds__(256) gemm_wmma(
    const bf16* __restrict__ A, const bf16* __restrict__ Bt,
    const float* __restrict__ bias, int M, int N, int K,
    int mode, int use_gelu,
    const float* __restrict__ res, float* __restrict__ outf,
    bf16* __restrict__ outb,
    bf16* __restrict__ qo, bf16* __restrict__ ko, bf16* __restrict__ vto)
{
    constexpr int PITCH = 40;                       // bf16 elems per row (80 B)
    __shared__ __align__(16) bf16 la[2][128 * PITCH];
    __shared__ __align__(16) bf16 lb[2][128 * PITCH];

    const int tid  = threadIdx.x;
    const int lane = tid & 31;
    const int warp = tid >> 5;
    const int wm = warp >> 1, wn = warp & 1;        // 4x2 wave grid over 128x128
    const int hi = lane >> 4, lx = lane & 15;
    const int kbase = hi * 8;
    const int m0 = blockIdx.x * 128, n0 = blockIdx.y * 128;

    const bf16* Abase = A  + (size_t)m0 * K;
    const bf16* Bbase = Bt + (size_t)n0 * K;

    // 128x32 tile copy: 512 uint4 / 256 threads = 2 per thread
    auto copy_tile = [&](const bf16* base, int k0, bf16* dst) {
#pragma unroll
        for (int r = 0; r < 2; ++r) {
            const int id  = tid + r * 256;
            const int row = id >> 2;
            const int col = (id & 3) * 8;
            copy16_g2l(base + (size_t)row * K + k0 + col, dst + row * PITCH + col);
        }
    };

    f32x8 acc[2][4];
#pragma unroll
    for (int i = 0; i < 2; ++i)
#pragma unroll
        for (int j = 0; j < 4; ++j) acc[i][j] = zero8();

    const int nk = K >> 5;
    copy_tile(Abase, 0, la[0]);
    copy_tile(Bbase, 0, lb[0]);
    wait_async_lds();
    __syncthreads();

    int cur = 0;
    for (int i = 0; i < nk; ++i) {
        if (i + 1 < nk) {                            // prefetch next k-tile
            copy_tile(Abase, (i + 1) << 5, la[cur ^ 1]);
            copy_tile(Bbase, (i + 1) << 5, lb[cur ^ 1]);
        }
        FragBF af[2], bfr[4];
#pragma unroll
        for (int mt = 0; mt < 2; ++mt) {
            const int row = wm * 32 + mt * 16 + lx;
            af[mt].q[0] = *reinterpret_cast<const uint4*>(&la[cur][row * PITCH + kbase]);
            af[mt].q[1] = *reinterpret_cast<const uint4*>(&la[cur][row * PITCH + 16 + kbase]);
        }
#pragma unroll
        for (int nt = 0; nt < 4; ++nt) {
            const int col = wn * 64 + nt * 16 + lx;
            bfr[nt].q[0] = *reinterpret_cast<const uint4*>(&lb[cur][col * PITCH + kbase]);
            bfr[nt].q[1] = *reinterpret_cast<const uint4*>(&lb[cur][col * PITCH + 16 + kbase]);
        }
#pragma unroll
        for (int mt = 0; mt < 2; ++mt)
#pragma unroll
            for (int nt = 0; nt < 4; ++nt)
                acc[mt][nt] = wmma_bf16(af[mt], bfr[nt], acc[mt][nt]);

        wait_async_lds();
        __syncthreads();
        cur ^= 1;
    }

    // epilogue
#pragma unroll
    for (int mt = 0; mt < 2; ++mt) {
#pragma unroll
        for (int nt = 0; nt < 4; ++nt) {
            const int gn = n0 + wn * 64 + nt * 16 + lx;
            const float bb = bias ? bias[gn] : 0.0f;
            int which = 0, hh = 0, hd = 0;
            if (mode == 2) {
                which = gn >> 10;
                int rem = gn & 1023;
                hh = rem >> 6;
                hd = rem & 63;
            }
#pragma unroll
            for (int i = 0; i < 8; ++i) {
                const int gm = m0 + wm * 32 + mt * 16 + i + 8 * hi;
                float v = acc[mt][nt][i] + bb;
                if (mode == 0) {
                    if (use_gelu) v = 0.5f * v * (1.0f + erff(v * 0.70710678f));
                    outb[(size_t)gm * N + gn] = (bf16)v;
                } else if (mode == 1) {
                    outf[(size_t)gm * N + gn] = res[(size_t)gm * N + gn] + v;
                } else {
                    const int bidx = gm >> 11, qpos = gm & 2047;
                    const int bh = bidx * NHEADS + hh;
                    if (which == 0)
                        qo[((size_t)bh * SEQ + qpos) * HDIM + hd] = (bf16)v;
                    else if (which == 1)
                        ko[((size_t)bh * SEQ + qpos) * HDIM + hd] = (bf16)v;
                    else
                        vto[((size_t)bh * HDIM + hd) * SEQ + qpos] = (bf16)v;
                }
            }
        }
    }
}

// ---------------------------------------------------------------------------
// Flash attention: one wave per (b, h, 16-row q-tile); kv step 32; online
// softmax; V pre-transposed [B,H,Hd,N] so all WMMA operands load contiguous-K.
// ---------------------------------------------------------------------------
__global__ void __launch_bounds__(256) flash_attn(
    const bf16* __restrict__ Q, const bf16* __restrict__ Km,
    const bf16* __restrict__ Vt, bf16* __restrict__ O)
{
    __shared__ __align__(16) bf16 pbuf[8 * 16 * 40];   // per-wave P scratch

    const int tid  = threadIdx.x;
    const int lane = tid & 31;
    const int warp = tid >> 5;
    const int hi = lane >> 4, lx = lane & 15;
    const int kbase = hi * 8;

    const int gw    = blockIdx.x * 8 + warp;
    const int qtile = gw & 127;          // SEQ/16 = 128 tiles
    const int bh    = gw >> 7;           // 0..B*H-1
    const int q0    = qtile * 16;

    const bf16* Qb = Q  + (size_t)bh * SEQ * HDIM;
    const bf16* Kb = Km + (size_t)bh * SEQ * HDIM;
    const bf16* Vb = Vt + (size_t)bh * HDIM * SEQ;
    bf16* pw = pbuf + warp * 16 * 40;

    // Q fragments (A-matrix, 16x64 = two 16x32 chunks)
    FragBF qa[2];
#pragma unroll
    for (int c = 0; c < 2; ++c) {
        const bf16* p = Qb + (size_t)(q0 + lx) * HDIM + c * 32 + kbase;
        qa[c].q[0] = *reinterpret_cast<const uint4*>(p);
        qa[c].q[1] = *reinterpret_cast<const uint4*>(p + 16);
    }

    f32x8 oacc[4];
#pragma unroll
    for (int t = 0; t < 4; ++t) oacc[t] = zero8();
    float mrow[8], lrow[8];
#pragma unroll
    for (int i = 0; i < 8; ++i) { mrow[i] = -3.0e38f; lrow[i] = 0.0f; }

    for (int j0 = 0; j0 < SEQ; j0 += 32) {
        // K^T fragments (B-matrix): col = kv index, elements = contiguous Hd
        FragBF kb[2][2];
#pragma unroll
        for (int nt = 0; nt < 2; ++nt) {
            const bf16* p = Kb + (size_t)(j0 + nt * 16 + lx) * HDIM;
#pragma unroll
            for (int c = 0; c < 2; ++c) {
                kb[nt][c].q[0] = *reinterpret_cast<const uint4*>(p + c * 32 + kbase);
                kb[nt][c].q[1] = *reinterpret_cast<const uint4*>(p + c * 32 + 16 + kbase);
            }
        }
        f32x8 s0 = zero8(), s1 = zero8();
        s0 = wmma_bf16(qa[0], kb[0][0], s0);
        s0 = wmma_bf16(qa[1], kb[0][1], s0);
        s1 = wmma_bf16(qa[0], kb[1][0], s1);
        s1 = wmma_bf16(qa[1], kb[1][1], s1);

        float p0[8], p1[8], mnew[8], alpha[8];
#pragma unroll
        for (int i = 0; i < 8; ++i) {
            float a = s0[i] * 0.125f;        // Hd^-0.5
            float b = s1[i] * 0.125f;
            p0[i] = a; p1[i] = b;
            float mc = fmaxf(a, b);
#pragma unroll
            for (int m = 1; m < 16; m <<= 1) mc = fmaxf(mc, __shfl_xor(mc, m, 32));
            mnew[i]  = fmaxf(mrow[i], mc);
            alpha[i] = __expf(mrow[i] - mnew[i]);
            mrow[i]  = mnew[i];
        }
#pragma unroll
        for (int i = 0; i < 8; ++i) {
            float a = __expf(p0[i] - mnew[i]);
            float b = __expf(p1[i] - mnew[i]);
            float rs = a + b;
#pragma unroll
            for (int m = 1; m < 16; m <<= 1) rs += __shfl_xor(rs, m, 32);
            lrow[i] = lrow[i] * alpha[i] + rs;
#pragma unroll
            for (int t = 0; t < 4; ++t) oacc[t][i] *= alpha[i];
            const int row = i + 8 * hi;        // C-layout row for this lane
            pw[row * 40 + lx]      = (bf16)a;  // kv cols j0..j0+15
            pw[row * 40 + 16 + lx] = (bf16)b;  // kv cols j0+16..j0+31
        }
        // P as A-fragment (16x32) from LDS (wave-private, LDS pipe in-order)
        FragBF pa;
        pa.q[0] = *reinterpret_cast<const uint4*>(&pw[lx * 40 + kbase]);
        pa.q[1] = *reinterpret_cast<const uint4*>(&pw[lx * 40 + 16 + kbase]);
        // V fragments (B-matrix): col = hd, elements = contiguous kv (V transposed)
        FragBF vb[4];
#pragma unroll
        for (int t = 0; t < 4; ++t) {
            const bf16* p = Vb + (size_t)(t * 16 + lx) * SEQ + j0 + kbase;
            vb[t].q[0] = *reinterpret_cast<const uint4*>(p);
            vb[t].q[1] = *reinterpret_cast<const uint4*>(p + 16);
        }
#pragma unroll
        for (int t = 0; t < 4; ++t) oacc[t] = wmma_bf16(pa, vb[t], oacc[t]);
    }

    // normalize and store to attn_out [B, N, D]
    float inv[8];
#pragma unroll
    for (int i = 0; i < 8; ++i) inv[i] = 1.0f / lrow[i];
    const int b_ = bh >> 4, h = bh & 15;
#pragma unroll
    for (int t = 0; t < 4; ++t) {
#pragma unroll
        for (int i = 0; i < 8; ++i) {
            const int q   = q0 + i + 8 * hi;
            const int col = h * 64 + t * 16 + lx;
            O[((size_t)b_ * SEQ + q) * EMBED + col] = (bf16)(oacc[t][i] * inv[i]);
        }
    }
}

// ---------------------------------------------------------------------------
// Host orchestration
// ---------------------------------------------------------------------------
extern "C" void kernel_launch(void* const* d_in, const int* in_sizes, int n_in,
                              void* d_out, int out_size, void* d_ws, size_t ws_size,
                              hipStream_t stream)
{
    const float* x     = (const float*)d_in[0];
    const float* w_qkv = (const float*)d_in[1];
    const float* b_qkv = (const float*)d_in[2];
    const float* w_out = (const float*)d_in[3];
    const float* b_out = (const float*)d_in[4];
    const float* w_fc1 = (const float*)d_in[5];
    const float* b_fc1 = (const float*)d_in[6];
    const float* w_fc2 = (const float*)d_in[7];
    const float* b_fc2 = (const float*)d_in[8];
    const float* ln1_g = (const float*)d_in[9];
    const float* ln1_b = (const float*)d_in[10];
    const float* ln2_g = (const float*)d_in[11];
    const float* ln2_b = (const float*)d_in[12];
    float* out = (float*)d_out;

    char* ws = (char*)d_ws;
    size_t off = 0;
    auto alloc = [&](size_t bytes) -> char* {
        char* p = ws + off;
        off += (bytes + 255) & ~(size_t)255;
        return p;
    };

    // weights stored transposed: [N,K] bf16
    bf16*  wqkv_t = (bf16*)alloc((size_t)3 * EMBED * EMBED * sizeof(bf16));
    bf16*  wout_t = (bf16*)alloc((size_t)EMBED * EMBED * sizeof(bf16));
    bf16*  wfc1_t = (bf16*)alloc((size_t)HID * EMBED * sizeof(bf16));
    bf16*  wfc2_t = (bf16*)alloc((size_t)EMBED * HID * sizeof(bf16));
    bf16*  h_b    = (bf16*)alloc((size_t)MROWS * EMBED * sizeof(bf16));
    bf16*  q_b    = (bf16*)alloc((size_t)MROWS * EMBED * sizeof(bf16));
    bf16*  k_b    = (bf16*)alloc((size_t)MROWS * EMBED * sizeof(bf16));
    bf16*  vt_b   = (bf16*)alloc((size_t)MROWS * EMBED * sizeof(bf16));
    bf16*  attn_b = (bf16*)alloc((size_t)MROWS * EMBED * sizeof(bf16));
    float* x1     = (float*)alloc((size_t)MROWS * EMBED * sizeof(float));
    bf16*  ffh_b  = (bf16*)alloc((size_t)MROWS * HID * sizeof(bf16));
    (void)ws_size; (void)in_sizes; (void)n_in; (void)out_size;

    // 1) weights fp32 [K,N] -> bf16 [N,K]
    {
        dim3 blk(32, 8);
        cvt_transpose_bf16<<<dim3((3 * EMBED) / 32, EMBED / 32), blk, 0, stream>>>(
            w_qkv, wqkv_t, EMBED, 3 * EMBED);
        cvt_transpose_bf16<<<dim3(EMBED / 32, EMBED / 32), blk, 0, stream>>>(
            w_out, wout_t, EMBED, EMBED);
        cvt_transpose_bf16<<<dim3(HID / 32, EMBED / 32), blk, 0, stream>>>(
            w_fc1, wfc1_t, EMBED, HID);
        cvt_transpose_bf16<<<dim3(EMBED / 32, HID / 32), blk, 0, stream>>>(
            w_fc2, wfc2_t, HID, EMBED);
    }

    // 2) LN1
    layernorm_bf16<<<MROWS, 256, 0, stream>>>(x, ln1_g, ln1_b, h_b);

    // 3) QKV GEMM with scatter epilogue (V stored transposed)
    {
        dim3 g(MROWS / 128, (3 * EMBED) / 128);
        gemm_wmma<<<g, 256, 0, stream>>>(h_b, wqkv_t, b_qkv, MROWS, 3 * EMBED, EMBED,
                                         /*mode=*/2, 0, nullptr, nullptr, nullptr,
                                         q_b, k_b, vt_b);
    }

    // 4) flash attention: B*H*(N/16) = 4096 waves / 8 per block
    flash_attn<<<512, 256, 0, stream>>>(q_b, k_b, vt_b, attn_b);

    // 5) output projection + residual -> x1 (f32)
    {
        dim3 g(MROWS / 128, EMBED / 128);
        gemm_wmma<<<g, 256, 0, stream>>>(attn_b, wout_t, b_out, MROWS, EMBED, EMBED,
                                         /*mode=*/1, 0, x, x1, nullptr,
                                         nullptr, nullptr, nullptr);
    }

    // 6) LN2
    layernorm_bf16<<<MROWS, 256, 0, stream>>>(x1, ln2_g, ln2_b, h_b);

    // 7) FC1 + erf-GELU -> bf16
    {
        dim3 g(MROWS / 128, HID / 128);
        gemm_wmma<<<g, 256, 0, stream>>>(h_b, wfc1_t, b_fc1, MROWS, HID, EMBED,
                                         /*mode=*/0, /*gelu=*/1, nullptr, nullptr, ffh_b,
                                         nullptr, nullptr, nullptr);
    }

    // 8) FC2 + residual -> d_out (f32)
    {
        dim3 g(MROWS / 128, EMBED / 128);
        gemm_wmma<<<g, 256, 0, stream>>>(ffh_b, wfc2_t, b_fc2, MROWS, EMBED, HID,
                                         /*mode=*/1, 0, x1, out, nullptr,
                                         nullptr, nullptr, nullptr);
    }
}
